// StripeLayer_57982058496028
// MI455X (gfx1250) — matrix operand order repaired and checked by reading, not compiled, not run on hardware
//
#include <hip/hip_runtime.h>
#include <hip/hip_bf16.h>

typedef _Float16 half_t;
typedef __attribute__((ext_vector_type(16))) _Float16 v16h;
typedef __attribute__((ext_vector_type(8)))  float    v8f;
typedef int gint4 __attribute__((vector_size(16)));   // matches builtin's V4i

#define BM 128
#define BN 128
#define BK 32
#define PAD 8

// 1/sqrt(1 + 1e-5)  (eval-mode BN with running stats 0/1)
#define BN_INV 0.9999950000374997f

#if __has_builtin(__builtin_amdgcn_global_load_async_to_lds_b128)
#define ASYNC_LDS 1
#else
#define ASYNC_LDS 0
#endif

// 16-byte global -> LDS copy; async DMA (ASYNCcnt) when available.
__device__ __forceinline__ void async_cp16(void* lds_dst, const void* gsrc) {
#if ASYNC_LDS
  __builtin_amdgcn_global_load_async_to_lds_b128(
      (__attribute__((address_space(1))) gint4*)gsrc,
      (__attribute__((address_space(3))) gint4*)lds_dst, 0, 0);
#else
  *(float4*)lds_dst = *(const float4*)gsrc;
#endif
}

__device__ __forceinline__ void wait_async_all() {
#if ASYNC_LDS
#if __has_builtin(__builtin_amdgcn_s_wait_asynccnt)
  __builtin_amdgcn_s_wait_asynccnt(0);
#else
  asm volatile("s_wait_asynccnt 0x0" ::: "memory");
#endif
#endif
}

template <int N>
__device__ __forceinline__ void wait_async_le() {
#if ASYNC_LDS
#if __has_builtin(__builtin_amdgcn_s_wait_asynccnt)
  __builtin_amdgcn_s_wait_asynccnt(N);
#else
  asm volatile("s_wait_asynccnt %0" ::"i"(N) : "memory");
#endif
#endif
}

__device__ __forceinline__ v8f wmma_f16(v16h a, v16h b, v8f c) {
  return __builtin_amdgcn_wmma_f32_16x16x32_f16(false, a, false, b, (short)0, c,
                                                false, false);
}

// A fragment 16(M) x 32(K), f16 source.  elem(m,k) = base[m*sm + k*sk]
__device__ __forceinline__ v16h frag_a_h(const half_t* base, int sm, int sk,
                                         int m0, int k0, int lane) {
  int m  = m0 + (lane & 15);
  int kb = k0 + ((lane >> 4) << 3);
  v16h f;
#pragma unroll
  for (int e = 0; e < 16; ++e) {
    int k = kb + ((e >> 3) << 4) + (((e >> 1) & 3) << 1) + (e & 1);
    f[e] = base[m * sm + k * sk];
  }
  return f;
}

// Same but f32 source (converted per element) — used for softmax probs.
__device__ __forceinline__ v16h frag_a_f(const float* base, int sm, int sk,
                                         int m0, int k0, int lane) {
  int m  = m0 + (lane & 15);
  int kb = k0 + ((lane >> 4) << 3);
  v16h f;
#pragma unroll
  for (int e = 0; e < 16; ++e) {
    int k = kb + ((e >> 3) << 4) + (((e >> 1) & 3) << 1) + (e & 1);
    f[e] = (half_t)base[m * sm + k * sk];
  }
  return f;
}

// B fragment 32(K) x 16(N).  elem(k,n) = base[k*sk + n*sn]
__device__ __forceinline__ v16h frag_b(const half_t* base, int sk, int sn,
                                       int k0, int n0, int lane) {
  int n  = n0 + (lane & 15);
  int kb = k0 + ((lane >> 4) << 3);
  v16h f;
#pragma unroll
  for (int e = 0; e < 16; ++e)
    f[e] = base[(kb + ((e >> 3) << 4) + (e & 7)) * sk + n * sn];
  return f;
}

// f32 -> f16 weight pre-conversion (runs once; weights stay hot in 192MB L2)
__global__ __launch_bounds__(256) void cvt_kernel(const float* __restrict__ s,
                                                  half_t* __restrict__ d,
                                                  int n) {
  int i = (blockIdx.x * 256 + threadIdx.x) * 4;
  if (i + 3 < n) {
    float4 v = *(const float4*)(s + i);
    d[i]     = (half_t)v.x;
    d[i + 1] = (half_t)v.y;
    d[i + 2] = (half_t)v.z;
    d[i + 3] = (half_t)v.w;
  }
}

// EPI modes
//  0: f16 out, relu((acc+bias)*g*inv + b)          (conv1)
//  1: f16 out, acc*alpha                            (q/k/v projections)
//  2: f32 out channel-major, relu(bn3(acc+bias)+x)  (conv3 + residual)
template <bool A_CHANMAJOR, int EPI>
__global__ __launch_bounds__(256) void gemm_kernel(
    const void* __restrict__ Ap, const half_t* __restrict__ Wp,
    const float* __restrict__ bias, const float* __restrict__ gam,
    const float* __restrict__ bet, const float* __restrict__ resid,
    void* __restrict__ outp, int M, int N, int K, int spatial, float alpha) {
  constexpr int ASZ = A_CHANMAJOR ? BK * (BM + PAD) : BM * (BK + PAD);
  __shared__ half_t lds_a[2][ASZ];
  __shared__ half_t lds_b[2][BN * (BK + PAD)];

  const int t    = threadIdx.x;
  const int wave = t >> 5;
  const int lane = t & 31;
  const int m0   = blockIdx.x * BM;
  const int n0   = blockIdx.y * BN;
  const int wm   = (wave & 3) * 32;
  const int wn   = (wave >> 2) * 64;

  const v8f vzero = {0.f, 0.f, 0.f, 0.f, 0.f, 0.f, 0.f, 0.f};
  v8f acc[2][4];
#pragma unroll
  for (int i = 0; i < 2; ++i)
#pragma unroll
    for (int j = 0; j < 4; ++j) acc[i][j] = vzero;

  auto stage = [&](int kt, int buf) {
    // B tile: f16 weights (N,K), 16B chunks, async to LDS
#pragma unroll
    for (int i = 0; i < 2; ++i) {
      int c  = i * 256 + t;        // 512 chunks of 16B
      int n  = c >> 2;             // 4 chunks per 32-f16 row
      int k8 = (c & 3) * 8;
      async_cp16(&lds_b[buf][n * (BK + PAD) + k8],
                 Wp + (size_t)(n0 + n) * K + kt + k8);
    }
    if constexpr (A_CHANMAJOR) {
      // conv1 input: f32 channel-major -> convert through VALU
      const float* A = (const float*)Ap;
      int bimg = m0 / spatial;
      int s0   = m0 % spatial;
      size_t abase = (size_t)bimg * K * spatial + (size_t)kt * spatial + s0;
#pragma unroll
      for (int i = 0; i < (BM * BK) / 256; ++i) {
        int e = i * 256 + t;
        int m = e & (BM - 1);
        int k = e >> 7;
        lds_a[buf][k * (BM + PAD) + m] =
            (half_t)A[abase + (size_t)k * spatial + m];
      }
    } else {
      // f16 row-major activations, async to LDS
      const half_t* A = (const half_t*)Ap;
#pragma unroll
      for (int i = 0; i < 2; ++i) {
        int c  = i * 256 + t;
        int m  = c >> 2;
        int k8 = (c & 3) * 8;
        async_cp16(&lds_a[buf][m * (BK + PAD) + k8],
                   A + (size_t)(m0 + m) * K + kt + k8);
      }
    }
  };

  stage(0, 0);
  int buf = 0;
  for (int kt = 0; kt < K; kt += BK, buf ^= 1) {
    const bool more = (kt + BK) < K;
    if (more) {
      stage(kt + BK, buf ^ 1);
      wait_async_le<A_CHANMAJOR ? 2 : 4>();  // next tile's issues may remain
    } else {
      wait_async_all();
    }
    __syncthreads();

    v16h bf[4];
#pragma unroll
    for (int j = 0; j < 4; ++j)
      bf[j] = frag_b(lds_b[buf], 1, BK + PAD, 0, wn + j * 16, lane);
#pragma unroll
    for (int i = 0; i < 2; ++i) {
      v16h af = A_CHANMAJOR
                    ? frag_a_h(lds_a[buf], 1, BM + PAD, wm + i * 16, 0, lane)
                    : frag_a_h(lds_a[buf], BK + PAD, 1, wm + i * 16, 0, lane);
#pragma unroll
      for (int j = 0; j < 4; ++j) acc[i][j] = wmma_f16(af, bf[j], acc[i][j]);
    }
    __syncthreads();
  }

  // ---- epilogue ----
#pragma unroll
  for (int i = 0; i < 2; ++i) {
#pragma unroll
    for (int j = 0; j < 4; ++j) {
      int n = n0 + wn + j * 16 + (lane & 15);
      float bs = 0.f, gs = 1.f, bb = 0.f;
      if constexpr (EPI != 1) {
        bs = bias[n];
        gs = gam[n] * BN_INV;
        bb = bet[n];
      }
#pragma unroll
      for (int r = 0; r < 8; ++r) {
        int m = m0 + wm + i * 16 + r + ((lane >> 4) << 3);
        float v = acc[i][j][r];
        if constexpr (EPI == 1) {
          ((half_t*)outp)[(size_t)m * N + n] = (half_t)(v * alpha);
        } else if constexpr (EPI == 0) {
          float y = (v + bs) * gs + bb;
          ((half_t*)outp)[(size_t)m * N + n] = (half_t)(y > 0.f ? y : 0.f);
        } else {
          int bimg = m / spatial, s = m % spatial;
          size_t idx = (size_t)bimg * N * spatial + (size_t)n * spatial + s;
          float y = (v + bs) * gs + bb + resid[idx];
          ((float*)outp)[idx] = y > 0.f ? y : 0.f;
        }
      }
    }
  }
}

// ---------------- attention ----------------
#define ATT_L 256
#define ATT_D 128
#define QB 64
#define KC 128
#define KVSTR (ATT_D + 8)
#define SSTR (ATT_L + 8)

__global__ __launch_bounds__(256) void attn_kernel(
    const half_t* __restrict__ q, const half_t* __restrict__ kbuf,
    const half_t* __restrict__ vbuf, const half_t* __restrict__ h1,
    const float* __restrict__ g2, const float* __restrict__ b2,
    half_t* __restrict__ out2) {
  __shared__ half_t lds_q[QB * KVSTR];    // 17 KB
  __shared__ half_t lds_kv[KC * KVSTR];   // 34 KB (K chunks, then V chunks)
  __shared__ float  lds_s[QB * SSTR];     // 66 KB logits / probs
  __shared__ float  lds_rsum[QB];

  const int t    = threadIdx.x;
  const int wave = t >> 5;
  const int lane = t & 31;
  const int bid  = blockIdx.x;
  const int qb   = bid & 3;
  const int head = (bid >> 2) & 3;
  const int bn   = bid >> 4;            // stripe-batch in [0,128)
  const int grow = bn * ATT_L + qb * QB;
  const int cbase = head * ATT_D;

  const int wm  = (wave & 3) * 16;
  const int wnb = (wave >> 2) * 64;
  const v8f vzero = {0.f, 0.f, 0.f, 0.f, 0.f, 0.f, 0.f, 0.f};

  // async-load Q block (64 x 128); q already scaled by d^-0.5
#pragma unroll
  for (int i = 0; i < 4; ++i) {
    int c  = i * 256 + t;       // 1024 chunks of 16B
    int r  = c >> 4;            // 16 chunks per 128-f16 row
    int k8 = (c & 15) * 8;
    async_cp16(&lds_q[r * KVSTR + k8],
               q + (size_t)(grow + r) * 512 + cbase + k8);
  }

  // ---- logits: S = Q K^T, K streamed in 2 chunks of 128 rows ----
  for (int ch = 0; ch < 2; ++ch) {
    __syncthreads();
#pragma unroll
    for (int i = 0; i < 8; ++i) {
      int c  = i * 256 + t;     // 2048 chunks
      int r  = c >> 4;
      int k8 = (c & 15) * 8;
      async_cp16(&lds_kv[r * KVSTR + k8],
                 kbuf + (size_t)(bn * ATT_L + ch * KC + r) * 512 + cbase + k8);
    }
    wait_async_all();
    __syncthreads();

    v8f sacc[4];
#pragma unroll
    for (int j = 0; j < 4; ++j) sacc[j] = vzero;
    for (int kk = 0; kk < ATT_D; kk += 32) {
      v16h af = frag_a_h(lds_q, KVSTR, 1, wm, kk, lane);
#pragma unroll
      for (int j = 0; j < 4; ++j) {
        // B = K^T: elem(k=d, n=key) = lds_kv[n*KVSTR + k]
        v16h bf = frag_b(lds_kv, 1, KVSTR, kk, wnb + j * 16, lane);
        sacc[j] = wmma_f16(af, bf, sacc[j]);
      }
    }
#pragma unroll
    for (int j = 0; j < 4; ++j)
#pragma unroll
      for (int r = 0; r < 8; ++r) {
        int m = wm + r + ((lane >> 4) << 3);
        int n = ch * KC + wnb + j * 16 + (lane & 15);
        lds_s[m * SSTR + n] = sacc[j][r];
      }
  }
  __syncthreads();

  // ---- softmax per row: keep e^(x-max) in f32, fold 1/sum into PV epilogue ----
  if (t < QB) {
    float* row = &lds_s[t * SSTR];
    float mx = row[0];
    for (int y = 1; y < ATT_L; ++y) mx = fmaxf(mx, row[y]);
    float sum = 0.f;
    for (int y = 0; y < ATT_L; ++y) {
      float e = __expf(row[y] - mx);
      sum += e;
      row[y] = e;
    }
    lds_rsum[t] = 1.0f / sum;
  }

  // ---- PV: attn = P V, V streamed through the same LDS buffer ----
  v8f oacc[4];
#pragma unroll
  for (int j = 0; j < 4; ++j) oacc[j] = vzero;
  for (int ch = 0; ch < 2; ++ch) {
    __syncthreads();
#pragma unroll
    for (int i = 0; i < 8; ++i) {
      int c  = i * 256 + t;
      int r  = c >> 4;
      int k8 = (c & 15) * 8;
      async_cp16(&lds_kv[r * KVSTR + k8],
                 vbuf + (size_t)(bn * ATT_L + ch * KC + r) * 512 + cbase + k8);
    }
    wait_async_all();
    __syncthreads();
    for (int kk = 0; kk < KC; kk += 32) {
      v16h af = frag_a_f(lds_s, SSTR, 1, wm, ch * KC + kk, lane);
#pragma unroll
      for (int j = 0; j < 4; ++j) {
        // B = V: elem(k=key, n=d) = lds_kv[k*KVSTR + n]
        v16h bf = frag_b(lds_kv, KVSTR, 1, kk, wnb + j * 16, lane);
        oacc[j] = wmma_f16(af, bf, oacc[j]);
      }
    }
  }
  __syncthreads();

  // ---- fused epilogue: (attn/sum + h1) -> BN2 -> ReLU -> f16 ----
#pragma unroll
  for (int j = 0; j < 4; ++j) {
    int nc  = wnb + j * 16 + (lane & 15);
    int col = cbase + nc;
    float gs = g2[col] * BN_INV;
    float bb = b2[col];
#pragma unroll
    for (int r = 0; r < 8; ++r) {
      int m  = wm + r + ((lane >> 4) << 3);
      int gm = grow + m;
      float o = oacc[j][r] * lds_rsum[m];
      float s = o + (float)h1[(size_t)gm * 512 + col];
      float y = s * gs + bb;
      out2[(size_t)gm * 512 + col] = (half_t)(y > 0.f ? y : 0.f);
    }
  }
}

extern "C" void kernel_launch(void* const* d_in, const int* in_sizes, int n_in,
                              void* d_out, int out_size, void* d_ws,
                              size_t ws_size, hipStream_t stream) {
  (void)in_sizes; (void)n_in; (void)out_size; (void)ws_size;
  const float* x   = (const float*)d_in[0];
  const float* w1  = (const float*)d_in[1];
  const float* b1  = (const float*)d_in[2];
  const float* g1  = (const float*)d_in[3];
  const float* be1 = (const float*)d_in[4];
  const float* wq  = (const float*)d_in[5];
  const float* wk  = (const float*)d_in[6];
  const float* wv  = (const float*)d_in[7];
  const float* g2  = (const float*)d_in[8];
  const float* be2 = (const float*)d_in[9];
  const float* w3  = (const float*)d_in[10];
  const float* b3  = (const float*)d_in[11];
  const float* g3  = (const float*)d_in[12];
  const float* be3 = (const float*)d_in[13];

  const int M = 32768, P = 512, C = 1024, S = 2048;
  char* ws = (char*)d_ws;
  const size_t seg = (size_t)M * P * sizeof(half_t);  // 32 MB
  half_t* h1 = (half_t*)ws;
  half_t* qb = (half_t*)(ws + seg);
  half_t* kb = (half_t*)(ws + 2 * seg);
  half_t* vb = (half_t*)(ws + 3 * seg);
  half_t* o2 = qb;  // each attn block reads its q rows before writing them
  // f16 weights after the activation segments (~3.5 MB)
  half_t* w1h = (half_t*)(ws + 4 * seg);
  half_t* wqh = w1h + (size_t)P * C;
  half_t* wkh = wqh + (size_t)P * P;
  half_t* wvh = wkh + (size_t)P * P;
  half_t* w3h = wvh + (size_t)P * P;

  dim3 blk(256);
  const float qscale = 0.08838834764831845f;  // 128^-0.5

  cvt_kernel<<<dim3((P * C) / 1024), blk, 0, stream>>>(w1, w1h, P * C);
  cvt_kernel<<<dim3((P * P) / 1024), blk, 0, stream>>>(wq, wqh, P * P);
  cvt_kernel<<<dim3((P * P) / 1024), blk, 0, stream>>>(wk, wkh, P * P);
  cvt_kernel<<<dim3((P * P) / 1024), blk, 0, stream>>>(wv, wvh, P * P);
  cvt_kernel<<<dim3((C * P) / 1024), blk, 0, stream>>>(w3, w3h, C * P);

  gemm_kernel<true, 0><<<dim3(M / BM, P / BN), blk, 0, stream>>>(
      x, w1h, b1, g1, be1, nullptr, h1, M, P, C, S, 1.f);
  gemm_kernel<false, 1><<<dim3(M / BM, P / BN), blk, 0, stream>>>(
      h1, wqh, nullptr, nullptr, nullptr, nullptr, qb, M, P, P, S, qscale);
  gemm_kernel<false, 1><<<dim3(M / BM, P / BN), blk, 0, stream>>>(
      h1, wkh, nullptr, nullptr, nullptr, nullptr, kb, M, P, P, S, 1.f);
  gemm_kernel<false, 1><<<dim3(M / BM, P / BN), blk, 0, stream>>>(
      h1, wvh, nullptr, nullptr, nullptr, nullptr, vb, M, P, P, S, 1.f);
  attn_kernel<<<dim3(2048), blk, 0, stream>>>(qb, kb, vb, h1, g2, be2, o2);
  gemm_kernel<false, 2><<<dim3(M / BM, C / BN), blk, 0, stream>>>(
      o2, w3h, b3, g3, be3, x, d_out, M, C, P, S, 1.f);
}
